// RGAT_9689446220165
// MI455X (gfx1250) — matrix operand order repaired and checked by reading, not compiled, not run on hardware
//
#include <hip/hip_runtime.h>
#include <hip/hip_bf16.h>

// ---------------------------------------------------------------------------
// RGAT forward for MI455X (gfx1250): bf16 WMMA GEMMs (software-pipelined
// global->LDS staging) + L2-resident atomic edge scatter. wave32 everywhere.
// ---------------------------------------------------------------------------

typedef __attribute__((ext_vector_type(16))) __bf16 v16bf;
typedef __attribute__((ext_vector_type(8)))  float  v8f;

#define NN 20000
#define EE 200000
#define HH 512
#define TM 128
#define TN 128
#define TK 32
#define LDT 40   // LDS row stride in bf16 (32 + 8 pad) -> 80 B, 16B aligned

// ---------------------------------------------------------------------------
// C[M x 512] = op(A[M x K] @ W[K x 512] + bias), op = identity or tanh.
// Block tile 128x128, 256 threads = 8 waves (4 along M x 2 along N),
// each wave: 2x4 tiles of 16x16 via v_wmma_f32_16x16x32_bf16.
// Software pipeline: global loads for step kt+1 issued before the WMMAs of
// step kt; LDS commit of the next step provides the load-wait point.
// ---------------------------------------------------------------------------
__global__ __launch_bounds__(256)
void rgat_gemm_wmma(const float* __restrict__ A, const float* __restrict__ W,
                    const float* __restrict__ bias, float* __restrict__ C,
                    int M, int K, int fuse_tanh) {
  const int N = HH;
  __shared__ __bf16 As[TM * LDT];
  __shared__ __bf16 Bs[TN * LDT];

  const int tid  = threadIdx.x;
  const int lane = tid & 31;
  const int wid  = tid >> 5;     // 0..7
  const int wm   = wid & 3;      // 32-row band
  const int wn   = wid >> 2;     // 64-col band
  const int m0   = blockIdx.x * TM;
  const int n0   = blockIdx.y * TN;
  const int lhalf = lane >> 4;   // 0/1
  const int lmod  = lane & 15;

  // Per-thread staging coordinates (fixed across k-steps).
  // A rows are CLAMPED, not predicated: M % 16 == 0, so out-of-range rows
  // only feed 16x16 output tiles that the epilogue never stores.
  int rA[4], cA[4], kW[4], nW[4];
  const float* gA[4];
  const float* gW[4];
  for (int i = 0; i < 4; ++i) {
    int q = tid + i * 256;                 // 0..1023
    rA[i] = q >> 3;                        // A-tile row
    cA[i] = (q & 7) << 2;                  // A-tile col (x4)
    int gm = m0 + rA[i]; if (gm >= M) gm = M - 1;
    gA[i] = A + (size_t)gm * K + cA[i];
    kW[i] = q >> 5;                        // W-tile k row (0..31)
    nW[i] = (q & 31) << 2;                 // W-tile n (x4)
    gW[i] = W + (size_t)kW[i] * N + n0 + nW[i];
  }

  v8f acc[2][4];
  for (int i = 0; i < 2; ++i)
    for (int j = 0; j < 4; ++j)
      for (int e = 0; e < 8; ++e) acc[i][j][e] = 0.0f;

  const int nsteps = K / TK;

  // prologue: stage tile 0 into registers
  float4 va[4], vw[4];
  for (int i = 0; i < 4; ++i) va[i] = *reinterpret_cast<const float4*>(gA[i]);
  for (int i = 0; i < 4; ++i) vw[i] = *reinterpret_cast<const float4*>(gW[i]);

  for (int kt = 0; kt < nsteps; ++kt) {
    // ---- commit staged registers to LDS (f32 -> bf16); W stored transposed
    for (int i = 0; i < 4; ++i) {
      __bf16* d = As + rA[i] * LDT + cA[i];
      d[0] = (__bf16)va[i].x; d[1] = (__bf16)va[i].y;
      d[2] = (__bf16)va[i].z; d[3] = (__bf16)va[i].w;
      Bs[(nW[i] + 0) * LDT + kW[i]] = (__bf16)vw[i].x;
      Bs[(nW[i] + 1) * LDT + kW[i]] = (__bf16)vw[i].y;
      Bs[(nW[i] + 2) * LDT + kW[i]] = (__bf16)vw[i].z;
      Bs[(nW[i] + 3) * LDT + kW[i]] = (__bf16)vw[i].w;
    }
    __syncthreads();

    // ---- prefetch next tile (VMEM in flight during DS + WMMA below)
    if (kt + 1 < nsteps) {
      for (int i = 0; i < 4; ++i)
        va[i] = *reinterpret_cast<const float4*>(gA[i] + (size_t)(kt + 1) * TK);
      for (int i = 0; i < 4; ++i)
        vw[i] = *reinterpret_cast<const float4*>(gW[i] + (size_t)(kt + 1) * TK * N);
    }

    // ---- fragments (contiguous k order, same for A and B)
    v16bf afrag[2], bfrag[4];
    for (int i = 0; i < 2; ++i) {
      // A 16x32 bf16: lane holds row lmod; K = lhalf*16 + {0..15} contiguous
      const __bf16* p = As + (wm * 32 + i * 16 + lmod) * LDT + lhalf * 16;
      reinterpret_cast<float4*>(&afrag[i])[0] = *reinterpret_cast<const float4*>(p);
      reinterpret_cast<float4*>(&afrag[i])[1] = *reinterpret_cast<const float4*>(p + 8);
    }
    for (int j = 0; j < 4; ++j) {
      // B 32x16 bf16: lane holds col lmod; K = lhalf*16 + {0..15} contiguous
      const __bf16* p = Bs + (wn * 64 + j * 16 + lmod) * LDT + lhalf * 16;
      reinterpret_cast<float4*>(&bfrag[j])[0] = *reinterpret_cast<const float4*>(p);
      reinterpret_cast<float4*>(&bfrag[j])[1] = *reinterpret_cast<const float4*>(p + 8);
    }
    for (int i = 0; i < 2; ++i)
      for (int j = 0; j < 4; ++j)
        acc[i][j] = __builtin_amdgcn_wmma_f32_16x16x32_bf16(
            false, afrag[i], false, bfrag[j], (short)0, acc[i][j], false, false);
    __syncthreads();
  }

  // ---- epilogue: +bias, optional tanh. C/D layout: lane -> (n=lmod, m=r+lhalf*8)
  for (int i = 0; i < 2; ++i) {
    int tileM = m0 + wm * 32 + i * 16;
    if (tileM >= M) continue;          // M % 16 == 0 -> whole-tile guard is exact
    for (int j = 0; j < 4; ++j) {
      int n = n0 + wn * 64 + j * 16 + lmod;
      float bv = bias ? bias[n] : 0.0f;
      for (int r = 0; r < 8; ++r) {
        int m = tileM + r + lhalf * 8;
        float v = acc[i][j][r] + bv;
        if (fuse_tanh) v = tanhf(v);
        C[(size_t)m * N + n] = v;
      }
    }
  }
}

// out[i] = dot(X[i,:512], w[:512]) ; one wave per row
__global__ __launch_bounds__(256)
void rgat_rowdot(const float* __restrict__ X, const float* __restrict__ w,
                 float* __restrict__ out, int M) {
  int row  = blockIdx.x * 8 + (threadIdx.x >> 5);
  int lane = threadIdx.x & 31;
  if (row >= M) return;
  const float* x = X + (size_t)row * HH;
  float s = 0.f;
  for (int j = 0; j < 16; ++j) s += x[lane + j * 32] * w[lane + j * 32];
  for (int off = 16; off; off >>= 1) s += __shfl_down(s, off, 32);
  if (lane == 0) out[row] = s;
}

// per edge: ex = exp(tanh(ssum[src]+tsum[tgt]+ab)); den[src] += ex
// (tanh output bounded in (-1,1) -> softmax max-subtraction unnecessary)
__global__ __launch_bounds__(256)
void rgat_edge1(const int* __restrict__ src, const int* __restrict__ tgt,
                const float* __restrict__ ssum, const float* __restrict__ tsum,
                const float* __restrict__ ab_p, float* __restrict__ ex,
                float* __restrict__ den, int E) {
  int i = blockIdx.x * 256 + threadIdx.x;
  if (i >= E) return;
  int s = src[i], t = tgt[i];
  float e = tanhf(ssum[s] + tsum[t] + ab_p[0]);
  float v = __expf(e);
  ex[i] = v;
  atomicAdd(den + s, v);
}

// acc = 0.5*(h + bias)  (folds GAT bias and the 0.5 mean into the accumulator)
__global__ __launch_bounds__(256)
void rgat_init_acc(const float* __restrict__ h, const float* __restrict__ bias,
                   float* __restrict__ acc, int M) {
  size_t i = (size_t)blockIdx.x * 256 + threadIdx.x;
  if (i >= (size_t)M * HH) return;
  acc[i] = 0.5f * (h[i] + bias[i & (HH - 1)]);
}

// per edge (one wave): acc[src,:] += 0.5*alpha * T[tgt,:]
__global__ __launch_bounds__(256)
void rgat_edge2(const int* __restrict__ src, const int* __restrict__ tgt,
                const float* __restrict__ ex, const float* __restrict__ den,
                const float* __restrict__ T, float* __restrict__ acc, int E) {
  int e    = blockIdx.x * 8 + (threadIdx.x >> 5);
  int lane = threadIdx.x & 31;
  if (e >= E) return;
  int s = src[e], t = tgt[e];
  float alpha = 0.5f * ex[e] / den[s];
  const float* tp = T + (size_t)t * HH;
  float* ap = acc + (size_t)s * HH;
  for (int j = 0; j < 4; ++j) {
    int k = j * 128 + lane * 4;
    float4 v = *reinterpret_cast<const float4*>(tp + k);
    atomicAdd(ap + k + 0, alpha * v.x);
    atomicAdd(ap + k + 1, alpha * v.y);
    atomicAdd(ap + k + 2, alpha * v.z);
    atomicAdd(ap + k + 3, alpha * v.w);
  }
}

// y = LN(x) or y = LN(tanh(x + sc)); one wave per 512-wide row
__global__ __launch_bounds__(256)
void rgat_ln(const float* __restrict__ x, const float* __restrict__ sc,
             const float* __restrict__ g, const float* __restrict__ b,
             float* __restrict__ y, int M) {
  int row  = blockIdx.x * 8 + (threadIdx.x >> 5);
  int lane = threadIdx.x & 31;
  if (row >= M) return;
  const float* xp = x + (size_t)row * HH;
  float v[16];
  float mean = 0.f;
  for (int j = 0; j < 16; ++j) {
    int k = lane + j * 32;
    float t = xp[k];
    if (sc) t = tanhf(t + sc[(size_t)row * HH + k]);
    v[j] = t; mean += t;
  }
  for (int off = 16; off; off >>= 1) mean += __shfl_xor(mean, off, 32);
  mean *= (1.f / HH);
  float var = 0.f;
  for (int j = 0; j < 16; ++j) { float d = v[j] - mean; var += d * d; }
  for (int off = 16; off; off >>= 1) var += __shfl_xor(var, off, 32);
  var *= (1.f / HH);
  float inv = rsqrtf(var + 1e-5f);
  float* yp = y + (size_t)row * HH;
  for (int j = 0; j < 16; ++j) {
    int k = lane + j * 32;
    yp[k] = (v[j] - mean) * inv * g[k] + b[k];
  }
}

// out[i] = sigmoid(dot(x[i], w) + b)
__global__ __launch_bounds__(256)
void rgat_cls(const float* __restrict__ x, const float* __restrict__ w,
              const float* __restrict__ bp, float* __restrict__ out, int M) {
  int row  = blockIdx.x * 8 + (threadIdx.x >> 5);
  int lane = threadIdx.x & 31;
  if (row >= M) return;
  const float* xp = x + (size_t)row * HH;
  float s = 0.f;
  for (int j = 0; j < 16; ++j) s += xp[lane + j * 32] * w[lane + j * 32];
  for (int off = 16; off; off >>= 1) s += __shfl_down(s, off, 32);
  if (lane == 0) out[row] = 1.f / (1.f + __expf(-(s + bp[0])));
}

// ---------------------------------------------------------------------------
extern "C" void kernel_launch(void* const* d_in, const int* in_sizes, int n_in,
                              void* d_out, int out_size, void* d_ws, size_t ws_size,
                              hipStream_t stream) {
  (void)in_sizes; (void)n_in; (void)out_size; (void)ws_size;

  const float* fu0 = (const float*)d_in[0];
  const float* fu1 = (const float*)d_in[2];
  const float* fi1 = (const float*)d_in[3];
  const float* fu2 = (const float*)d_in[4];
  const float* fi2 = (const float*)d_in[5];
  const int*   eUI0 = (const int*)d_in[6];
  const int*   eUI1 = (const int*)d_in[8];
  const int*   eIU1 = (const int*)d_in[9];
  const float* Wu = (const float*)d_in[10];
  const float* bu = (const float*)d_in[11];
  const float* Wi = (const float*)d_in[12];
  const float* bi = (const float*)d_in[13];
  const float* gfW   = (const float*)d_in[14];  // [2,512,512]
  const float* gfb   = (const float*)d_in[15];  // [2,512]
  const float* gaw   = (const float*)d_in[16];  // [2,1024]
  const float* gab   = (const float*)d_in[17];  // [2]
  const float* gbias = (const float*)d_in[18];  // [2,512]
  const float* prepW = (const float*)d_in[19];
  const float* prepb = (const float*)d_in[20];
  const float* dnnW  = (const float*)d_in[21];  // [2,2,512,512]
  const float* dnnb  = (const float*)d_in[22];  // [2,2,512]
  const float* lng   = (const float*)d_in[23];
  const float* lnb   = (const float*)d_in[24];
  const float* rg    = (const float*)d_in[25];  // [2,512]
  const float* rb    = (const float*)d_in[26];
  const float* clsW  = (const float*)d_in[27];  // [512]
  const float* clsb  = (const float*)d_in[28];  // [1]

  const size_t BIG = (size_t)NN * HH;           // 10.24M floats
  float* ws = (float*)d_ws;
  float* B0 = ws + 0 * BIG;
  float* B1 = ws + 1 * BIG;
  float* B2 = ws + 2 * BIG;
  float* B3 = ws + 3 * BIG;
  float* B4 = ws + 4 * BIG;
  float* B5 = ws + 5 * BIG;
  float* B6 = ws + 6 * BIG;
  float* ssum = ws + 7 * BIG;
  float* tsum = ssum + 20480;
  float* den  = tsum + 20480;
  float* EX   = den + 20480;

  const dim3 gG((NN + TM - 1) / TM, HH / TN);   // 157 x 4
  const int  RB = NN / 8;                        // 2500 row-wave blocks
  const int  E1B = (EE + 255) / 256;             // 782
  const int  E2B = EE / 8;                       // 25000
  const int  EWB = (int)(BIG / 256);             // 40000 elementwise blocks

  auto gat = [&](int r, const float* head, const float* tail, const int* edges,
                 float* acc) {
    rgat_gemm_wmma<<<gG, 256, 0, stream>>>(head, gfW + (size_t)r * HH * HH,
                                           gfb + r * HH, B4, NN, HH, 0);  // s
    rgat_gemm_wmma<<<gG, 256, 0, stream>>>(tail, gfW + (size_t)r * HH * HH,
                                           gfb + r * HH, B5, NN, HH, 0);  // t
    rgat_rowdot<<<RB, 256, 0, stream>>>(B4, gaw + (size_t)r * 2 * HH, ssum, NN);
    rgat_rowdot<<<RB, 256, 0, stream>>>(B5, gaw + (size_t)r * 2 * HH + HH, tsum, NN);
    hipMemsetAsync(den, 0, NN * sizeof(float), stream);
    rgat_edge1<<<E1B, 256, 0, stream>>>(edges, edges + EE, ssum, tsum, gab + r,
                                        EX, den, EE);
    rgat_init_acc<<<EWB, 256, 0, stream>>>(head, gbias + (size_t)r * HH, acc, NN);
    rgat_edge2<<<E2B, 256, 0, stream>>>(edges, edges + EE, EX, den, B5, acc, EE);
  };

  // hop-2 transforms
  rgat_gemm_wmma<<<gG, 256, 0, stream>>>(fu2, Wu, bu, B0, NN, 512, 0);  // su
  rgat_gemm_wmma<<<gG, 256, 0, stream>>>(fi2, Wi, bi, B1, NN, 256, 0);  // si
  // hop-1 transforms
  rgat_gemm_wmma<<<gG, 256, 0, stream>>>(fu1, Wu, bu, B2, NN, 512, 0);  // hu
  rgat_gemm_wmma<<<gG, 256, 0, stream>>>(fi1, Wi, bi, B3, NN, 256, 0);  // hi
  // hop-1 GATs (both read the OLD su/si)
  gat(0, /*head=*/B2, /*tail=*/B1, eUI1, /*acc->*/B6);  // su_new in B6
  gat(1, /*head=*/B3, /*tail=*/B0, eIU1, /*acc->*/B2);  // si_new in B2
  // hop-0
  rgat_gemm_wmma<<<gG, 256, 0, stream>>>(fu0, Wu, bu, B0, NN, 512, 0);  // hu0
  gat(0, /*head=*/B0, /*tail=*/B2, eUI0, /*acc->*/B1);  // su_final in B1

  // Res_DNN head
  rgat_gemm_wmma<<<gG, 256, 0, stream>>>(B1, prepW, prepb, B0, NN, 512, 0); // x
  for (int r = 0; r < 2; ++r) {
    rgat_gemm_wmma<<<gG, 256, 0, stream>>>(B0, dnnW + (size_t)(r * 2 + 0) * HH * HH,
                                           dnnb + (r * 2 + 0) * HH, B3, NN, 512, 1);
    rgat_ln<<<RB, 256, 0, stream>>>(B3, nullptr, lng + (r * 2 + 0) * HH,
                                    lnb + (r * 2 + 0) * HH, B4, NN);
    rgat_gemm_wmma<<<gG, 256, 0, stream>>>(B4, dnnW + (size_t)(r * 2 + 1) * HH * HH,
                                           dnnb + (r * 2 + 1) * HH, B3, NN, 512, 1);
    rgat_ln<<<RB, 256, 0, stream>>>(B3, nullptr, lng + (r * 2 + 1) * HH,
                                    lnb + (r * 2 + 1) * HH, B5, NN);
    // x = LN(tanh(sc + x)), sc = B0 (unchanged through the block), in-place
    rgat_ln<<<RB, 256, 0, stream>>>(B5, B0, rg + r * HH, rb + r * HH, B0, NN);
  }
  rgat_cls<<<RB, 256, 0, stream>>>(B0, clsW, clsb, (float*)d_out, NN);
}
